// Network_Whole_23115513987554
// MI455X (gfx1250) — matrix operand order repaired and checked by reading, not compiled, not run on hardware
//
#include <hip/hip_runtime.h>
#include <hip/hip_bf16.h>
#include <cstdint>

#define DEVI __device__ __forceinline__

static constexpr int Bn   = 32;
static constexpr int Np   = 2048;
static constexpr int KNN  = 20;
static constexpr int Mcub = 16;
static constexpr int EMBD = 1024;
static constexpr int ZDIM = 128;
static constexpr int ATTD = 64;

typedef _Float16 h16 __attribute__((ext_vector_type(16)));
typedef _Float16 h8  __attribute__((ext_vector_type(8)));
typedef float    f8  __attribute__((ext_vector_type(8)));
typedef int      v4i __attribute__((ext_vector_type(4)));

// ---- gfx1250 async global->LDS path (guarded; falls back to sync staging) ----
#if defined(__has_builtin)
#  if __has_builtin(__builtin_amdgcn_global_load_async_to_lds_b128)
#    define HAS_ASYNC 1
#  endif
#endif
#ifndef HAS_ASYNC
#  define HAS_ASYNC 0
#endif

DEVI void async_b128(const _Float16* g, _Float16* l){
#if HAS_ASYNC
  __builtin_amdgcn_global_load_async_to_lds_b128(
      (__attribute__((address_space(1))) v4i*)g,
      (__attribute__((address_space(3))) v4i*)l, 0, 0);
#else
  *(h8*)l = *(const h8*)g;
#endif
}
DEVI void wait_async_le2(){
#if HAS_ASYNC
#  if __has_builtin(__builtin_amdgcn_s_wait_asynccnt)
  __builtin_amdgcn_s_wait_asynccnt(2);
#  else
  asm volatile("s_wait_asynccnt 0x2" ::: "memory");
#  endif
#endif
}
DEVI void wait_async_le0(){
#if HAS_ASYNC
#  if __has_builtin(__builtin_amdgcn_s_wait_asynccnt)
  __builtin_amdgcn_s_wait_asynccnt(0);
#  else
  asm volatile("s_wait_asynccnt 0x0" ::: "memory");
#  endif
#endif
}

// ---------------- workspace layout (bytes) ----------------
static constexpr size_t ALGN(size_t x){ return (x + 255) & ~size_t(255); }
static constexpr size_t OFF_IDX    = 0;
static constexpr size_t OFF_XPER   = ALGN(OFF_IDX    + sizeof(int)      * (size_t)Bn*Np*KNN);
static constexpr size_t OFF_C3H    = ALGN(OFF_XPER   + sizeof(_Float16) * (size_t)Bn*Np*128);
static constexpr size_t OFF_WQH    = ALGN(OFF_C3H    + sizeof(_Float16) * (size_t)EMBD*128);
static constexpr size_t OFF_C2AH   = ALGN(OFF_WQH    + sizeof(_Float16) * (size_t)ATTD*128);
static constexpr size_t OFF_C2BH   = ALGN(OFF_C2AH   + sizeof(_Float16) * (size_t)64*128);
static constexpr size_t OFF_STATS3 = ALGN(OFF_C2BH   + sizeof(_Float16) * (size_t)64*64);
static constexpr size_t OFF_MS3    = ALGN(OFF_STATS3 + sizeof(float)*EMBD*2);
static constexpr size_t OFF_STATSE = ALGN(OFF_MS3    + sizeof(float)*EMBD*2);   // 4 stages * 64ch * {sum,sq}
static constexpr size_t OFF_MSE    = ALGN(OFF_STATSE + sizeof(float)*4*64*2);
static constexpr size_t OFF_XG     = ALGN(OFF_MSE    + sizeof(float)*4*64*2);
static constexpr size_t OFF_Z      = ALGN(OFF_XG     + sizeof(float)*(size_t)Bn*EMBD);
static constexpr size_t OFF_HEADS  = ALGN(OFF_Z      + sizeof(float)*(size_t)Bn*ZDIM);
static constexpr size_t OFF_KC     = ALGN(OFF_HEADS  + sizeof(float)*(size_t)Bn*Mcub*15);
static constexpr size_t OFF_Q      = ALGN(OFF_KC     + sizeof(float)*(size_t)Bn*Mcub*ATTD);

// ---------------- output layout (floats) ----------------
static constexpr size_t OUT_VERTS  = 0;
static constexpr size_t OUT_ASSIGN = OUT_VERTS  + (size_t)Bn*Mcub*8*3;   // 12288
static constexpr size_t OUT_EXIST  = OUT_ASSIGN + (size_t)Bn*Np*Mcub;    // +1048576
static constexpr size_t OUT_MU     = OUT_EXIST  + (size_t)Bn*Mcub;
static constexpr size_t OUT_LV     = OUT_MU     + (size_t)Bn*ZDIM;

__constant__ float CUBEV[8][3] = {
  {-1,-1,-1},{-1,-1, 1},{-1, 1,-1},{-1, 1, 1},
  { 1,-1,-1},{ 1,-1, 1},{ 1, 1,-1},{ 1, 1, 1}};

DEVI float lrelu(float x){ return x >= 0.f ? x : 0.2f*x; }

// ================= kNN (tiled through LDS) =================
__global__ void knn_kernel(const float* __restrict__ pc, int* __restrict__ idxb){
  __shared__ float sp[128*3];
  const int b = blockIdx.y;
  const int n = blockIdx.x*128 + threadIdx.x;
  const float xi0 = pc[((size_t)(b*Np+n))*3+0];
  const float xi1 = pc[((size_t)(b*Np+n))*3+1];
  const float xi2 = pc[((size_t)(b*Np+n))*3+2];
  float bd[KNN]; int bi[KNN];
  #pragma unroll
  for (int k=0;k<KNN;k++){ bd[k] = -3.0e38f; bi[k] = 0; }
  for (int j0=0;j0<Np;j0+=128){
    __syncthreads();
    sp[threadIdx.x*3+0] = pc[((size_t)(b*Np+j0+threadIdx.x))*3+0];
    sp[threadIdx.x*3+1] = pc[((size_t)(b*Np+j0+threadIdx.x))*3+1];
    sp[threadIdx.x*3+2] = pc[((size_t)(b*Np+j0+threadIdx.x))*3+2];
    __syncthreads();
    for (int jj=0;jj<128;jj++){
      float dx = xi0 - sp[jj*3+0];
      float dy = xi1 - sp[jj*3+1];
      float dz = xi2 - sp[jj*3+2];
      float nd = -(dx*dx + dy*dy + dz*dz);   // neg squared distance
      if (nd > bd[KNN-1]){
        int pos = KNN-1;
        while (pos > 0 && bd[pos-1] < nd){
          bd[pos] = bd[pos-1]; bi[pos] = bi[pos-1]; pos--;
        }
        bd[pos] = nd; bi[pos] = j0 + jj;
      }
    }
  }
  for (int k=0;k<KNN;k++) idxb[((size_t)(b*Np+n))*KNN + k] = bi[k];
}

// ================= EdgeConv stage 1 (CIN=6, scalar, recompute 2-pass BN) ========
DEVI void build_feat6(float* f, const float* pc, const int* idxb, int b, int n, int k){
  const int j = idxb[((size_t)(b*Np+n))*KNN + k];
  #pragma unroll
  for (int c=0;c<3;c++){
    float ce = pc[((size_t)(b*Np+n))*3+c];
    float nb = pc[((size_t)(b*Np+j))*3+c];
    f[c] = nb - ce; f[3+c] = ce;
  }
}

__global__ void ec1_stats_a(const float* __restrict__ pc, const int* __restrict__ idxb,
                            const float* __restrict__ Wa, float* __restrict__ stats){
  __shared__ float red[128];
  for (int t=threadIdx.x;t<128;t+=blockDim.x) red[t]=0.f;
  __syncthreads();
  const long long p = (long long)blockIdx.x*blockDim.x + threadIdx.x;
  if (p < (long long)Bn*Np*KNN){
    int b = (int)(p / ((long long)Np*KNN));
    int r = (int)(p % ((long long)Np*KNN));
    int n = r / KNN, k = r % KNN;
    float f[6];
    build_feat6(f, pc, idxb, b, n, k);
    for (int o=0;o<64;o++){
      float a = 0.f;
      #pragma unroll
      for (int c=0;c<6;c++) a += Wa[o*6+c]*f[c];
      atomicAdd(&red[o], a);
      atomicAdd(&red[64+o], a*a);
    }
  }
  __syncthreads();
  if (threadIdx.x < 64){
    atomicAdd(&stats[2*threadIdx.x],   red[threadIdx.x]);
    atomicAdd(&stats[2*threadIdx.x+1], red[64+threadIdx.x]);
  }
}

__global__ void ec1_stats_b(const float* __restrict__ pc, const int* __restrict__ idxb,
                            const float* __restrict__ Wa, const float* __restrict__ msA,
                            const float* __restrict__ gA, const float* __restrict__ bA,
                            const float* __restrict__ Wb, float* __restrict__ stats){
  __shared__ float red[128];
  for (int t=threadIdx.x;t<128;t+=blockDim.x) red[t]=0.f;
  __syncthreads();
  const long long p = (long long)blockIdx.x*blockDim.x + threadIdx.x;
  if (p < (long long)Bn*Np*KNN){
    int b = (int)(p / ((long long)Np*KNN));
    int r = (int)(p % ((long long)Np*KNN));
    int n = r / KNN, k = r % KNN;
    float f[6];
    build_feat6(f, pc, idxb, b, n, k);
    float ha[64];
    for (int o=0;o<64;o++){
      float a = 0.f;
      #pragma unroll
      for (int c=0;c<6;c++) a += Wa[o*6+c]*f[c];
      ha[o] = lrelu((a - msA[2*o])*msA[2*o+1]*gA[o] + bA[o]);
    }
    for (int o2=0;o2<64;o2++){
      float a = 0.f;
      for (int c=0;c<64;c++) a += Wb[o2*64+c]*ha[c];
      atomicAdd(&red[o2], a);
      atomicAdd(&red[64+o2], a*a);
    }
  }
  __syncthreads();
  if (threadIdx.x < 64){
    atomicAdd(&stats[2*threadIdx.x],   red[threadIdx.x]);
    atomicAdd(&stats[2*threadIdx.x+1], red[64+threadIdx.x]);
  }
}

// stage-1 final: writes f16 channels 0..63 of x_per (point-major)
__global__ void ec1_final(const float* __restrict__ pc, const int* __restrict__ idxb,
                          const float* __restrict__ Wa, const float* __restrict__ msA,
                          const float* __restrict__ gA, const float* __restrict__ bA,
                          const float* __restrict__ Wb, const float* __restrict__ msB,
                          const float* __restrict__ gB, const float* __restrict__ bB,
                          _Float16* __restrict__ xp){
  const int p = blockIdx.x*blockDim.x + threadIdx.x;
  if (p >= Bn*Np) return;
  const int b = p / Np, n = p % Np;
  float acc[64];
  for (int o=0;o<64;o++) acc[o] = -3.0e38f;
  for (int k=0;k<KNN;k++){
    float f[6];
    build_feat6(f, pc, idxb, b, n, k);
    float ha[64];
    for (int o=0;o<64;o++){
      float a = 0.f;
      #pragma unroll
      for (int c=0;c<6;c++) a += Wa[o*6+c]*f[c];
      ha[o] = lrelu((a - msA[2*o])*msA[2*o+1]*gA[o] + bA[o]);
    }
    for (int o2=0;o2<64;o2++){
      float a = 0.f;
      for (int c=0;c<64;c++) a += Wb[o2*64+c]*ha[c];
      a = lrelu((a - msB[2*o2])*msB[2*o2+1]*gB[o2] + bB[o2]);
      acc[o2] = fmaxf(acc[o2], a);
    }
  }
  for (int o=0;o<64;o++) xp[(size_t)p*128 + o] = (_Float16)acc[o];
}

__global__ void finalize_stats(const float* __restrict__ stats, float* __restrict__ ms,
                               int C, float invN){
  const int t = blockIdx.x*blockDim.x + threadIdx.x;
  if (t < C){
    float mean = stats[2*t]*invN;
    float var  = stats[2*t+1]*invN - mean*mean;
    ms[2*t]   = mean;
    ms[2*t+1] = rsqrtf(var + 1e-5f);
  }
}

__global__ void cvt_half(const float* __restrict__ src, _Float16* __restrict__ dst,
                         int nElem, float scale){
  const int t = blockIdx.x*blockDim.x + threadIdx.x;
  if (t < nElem) dst[t] = (_Float16)(src[t]*scale);
}

// ================= WMMA fragment loader (A: 16x32 f16, row = lane&15) ============
DEVI h16 load_a_frag(const _Float16* ap, int h){
  h8 lo = *(const h8*)(ap + h*16);
  h8 hi = *(const h8*)(ap + h*16 + 8);
  return __builtin_shufflevector(lo, hi, 0,1,2,3,4,5,6,7,8,9,10,11,12,13,14,15);
}

// ================= EdgeConv stage 2 (CIN=128) — WMMA through LDS =================
// One block per point (b,n): 20 edges padded to 32 rows; 8 waves = {4 o-subtiles} x {2 row-subtiles}.
// MODE 0: conv2a -> BN stats.  MODE 1: conv2a->bnA->lrelu->conv2b -> BN stats.
// MODE 2: full chain -> max over k -> write f16 channels 64..127 of x_per.
template<int MODE>
__global__ void ec2_wmma(const _Float16* __restrict__ xp, const int* __restrict__ idxb,
                         const _Float16* __restrict__ wa, const _Float16* __restrict__ wb,
                         const float* __restrict__ msA, const float* __restrict__ gA,
                         const float* __restrict__ bA, const float* __restrict__ msB,
                         const float* __restrict__ gB, const float* __restrict__ bB,
                         float* __restrict__ stats, _Float16* __restrict__ xpw){
  __shared__ _Float16 sfeat[32*128];
  __shared__ _Float16 sha[32*64];
  __shared__ float    sout[32*64];
  __shared__ float    ssum[64], ssq[64];
  const int bn  = blockIdx.x;             // b*Np + n
  const int tid = threadIdx.x;
  if (MODE != 2 && tid < 64){ ssum[tid] = 0.f; ssq[tid] = 0.f; }

  // ---- stage edge features (32 rows x 128 ch) into LDS as f16 ----
  {
    const int k = tid >> 3, cg = tid & 7;       // 32 rows, 8 chunks of 16 ch
    _Float16* dst = sfeat + k*128 + cg*16;
    if (k < KNN){
      const int b = bn / Np;
      const int j = idxb[(size_t)bn*KNN + k];
      const _Float16* ce = xp + (size_t)bn*128;
      const _Float16* nb = xp + ((size_t)(b*Np + j))*128;
      if (cg < 4){
        const int c0 = cg*16;
        #pragma unroll
        for (int c=0;c<16;c++)
          dst[c] = (_Float16)((float)nb[c0+c] - (float)ce[c0+c]);
      } else {
        const int c0 = (cg-4)*16;
        #pragma unroll
        for (int c=0;c<16;c++) dst[c] = ce[c0+c];
      }
    } else {
      #pragma unroll
      for (int c=0;c<16;c++) dst[c] = (_Float16)0.f;
    }
  }
  __syncthreads();

  const int w = tid >> 5, lane = tid & 31;
  const int osub = w & 3, rsub = w >> 2;
  const int r = lane & 15, h = lane >> 4;
  const int och = osub*16 + r;

  // ---- conv2a: 16x16 tile, K=128 in 4 WMMA steps (A from LDS, B from global) ----
  f8 c = {0.f,0.f,0.f,0.f,0.f,0.f,0.f,0.f};
  #pragma unroll
  for (int kk=0;kk<4;kk++){
    h16 av = load_a_frag(sfeat + (rsub*16 + r)*128 + kk*32, h);
    h16 bv = *(const h16*)(wa + (size_t)och*128 + kk*32 + h*16);
    c = __builtin_amdgcn_wmma_f32_16x16x32_f16(false, av, false, bv,
                                               (short)0, c, false, false);
  }

  if (MODE == 0){
    float sum = 0.f, sq = 0.f;
    #pragma unroll
    for (int i=0;i<8;i++){
      const int k = rsub*16 + i + h*8;
      const float v = (k < KNN) ? c[i] : 0.f;
      sum += v; sq += v*v;
    }
    sum += __shfl_xor(sum, 16, 32);
    sq  += __shfl_xor(sq,  16, 32);
    __syncthreads();
    if (lane < 16){ atomicAdd(&ssum[och], sum); atomicAdd(&ssq[och], sq); }
    __syncthreads();
    if (tid < 64){
      atomicAdd(&stats[2*tid],   ssum[tid]);
      atomicAdd(&stats[2*tid+1], ssq[tid]);
    }
    return;
  }

  // ---- bnA + lrelu -> LDS f16 (rows x 64) ----
  {
    const float mean = msA[2*och], inv = msA[2*och+1];
    const float gg = gA[och], bb = bA[och];
    #pragma unroll
    for (int i=0;i<8;i++){
      float y = (c[i]-mean)*inv*gg + bb;
      y = y >= 0.f ? y : 0.2f*y;
      const int row = rsub*16 + i + h*8;
      sha[row*64 + och] = (_Float16)y;
    }
  }
  __syncthreads();

  // ---- conv2b: K=64 in 2 WMMA steps ----
  f8 c2 = {0.f,0.f,0.f,0.f,0.f,0.f,0.f,0.f};
  #pragma unroll
  for (int kk=0;kk<2;kk++){
    h16 av = load_a_frag(sha + (rsub*16 + r)*64 + kk*32, h);
    h16 bv = *(const h16*)(wb + (size_t)och*64 + kk*32 + h*16);
    c2 = __builtin_amdgcn_wmma_f32_16x16x32_f16(false, av, false, bv,
                                                (short)0, c2, false, false);
  }

  if (MODE == 1){
    float sum = 0.f, sq = 0.f;
    #pragma unroll
    for (int i=0;i<8;i++){
      const int k = rsub*16 + i + h*8;
      const float v = (k < KNN) ? c2[i] : 0.f;
      sum += v; sq += v*v;
    }
    sum += __shfl_xor(sum, 16, 32);
    sq  += __shfl_xor(sq,  16, 32);
    if (lane < 16){ atomicAdd(&ssum[och], sum); atomicAdd(&ssq[och], sq); }
    __syncthreads();
    if (tid < 64){
      atomicAdd(&stats[2*tid],   ssum[tid]);
      atomicAdd(&stats[2*tid+1], ssq[tid]);
    }
    return;
  }

  // ---- MODE 2: bnB + lrelu -> LDS, max over valid k, write x_per[64..127] ----
  {
    const float mean = msB[2*och], inv = msB[2*och+1];
    const float gg = gB[och], bb = bB[och];
    #pragma unroll
    for (int i=0;i<8;i++){
      float y = (c2[i]-mean)*inv*gg + bb;
      y = y >= 0.f ? y : 0.2f*y;
      const int row = rsub*16 + i + h*8;
      sout[row*64 + och] = y;
    }
  }
  __syncthreads();
  if (tid < 64){
    float m = -3.0e38f;
    for (int k=0;k<KNN;k++) m = fmaxf(m, sout[k*64 + tid]);
    xpw[(size_t)bn*128 + 64 + tid] = (_Float16)m;
  }
}

// ================= c3 GEMM: stats pass (WMMA, async double-buffered A tiles) =====
__global__ void c3_stats(const _Float16* __restrict__ xp, const _Float16* __restrict__ wh,
                         float* __restrict__ stats){
  __shared__ _Float16 sfA[2][32*128];       // double-buffered A tile (8 KB each)
  __shared__ float ssum[64], ssq[64];
  const int tid = threadIdx.x;
  if (tid < 64){ ssum[tid]=0.f; ssq[tid]=0.f; }
  const int b = blockIdx.y, ot = blockIdx.x;
  const int w = tid >> 5, lane = tid & 31;
  const int osub = w & 3, nh = w >> 2;
  const int r = lane & 15, h = lane >> 4;
  // staging: 32 rows x 16 b128-chunks; 8 threads/row, 2 chunks each
  const int lrow = tid >> 3, lch = tid & 7;
  const _Float16* gbase = xp + (size_t)b*Np*128;
  h16 bf[4];
  #pragma unroll
  for (int kk=0;kk<4;kk++)
    bf[kk] = *(const h16*)(wh + ((size_t)(ot*64 + osub*16 + r))*128 + kk*32 + h*16);
  float sum = 0.f, sq = 0.f;
  constexpr int NT = Np/32;
  {
    const _Float16* gp = gbase + (size_t)lrow*128 + lch*8;
    _Float16* lp = &sfA[0][lrow*128 + lch*8];
    async_b128(gp, lp); async_b128(gp + 64, lp + 64);
  }
  for (int t=0;t<NT;t++){
    if (t+1 < NT){
      const _Float16* gp = gbase + (size_t)((t+1)*32 + lrow)*128 + lch*8;
      _Float16* lp = &sfA[(t+1)&1][lrow*128 + lch*8];
      async_b128(gp, lp); async_b128(gp + 64, lp + 64);
      wait_async_le2();          // previous tile complete (in-order async loads)
    } else {
      wait_async_le0();
    }
    __syncthreads();
    const _Float16* tile = sfA[t&1];
    f8 c = {0.f,0.f,0.f,0.f,0.f,0.f,0.f,0.f};
    #pragma unroll
    for (int kk=0;kk<4;kk++){
      h16 av = load_a_frag(tile + (nh*16 + r)*128 + kk*32, h);
      c = __builtin_amdgcn_wmma_f32_16x16x32_f16(false, av, false, bf[kk],
                                                 (short)0, c, false, false);
    }
    #pragma unroll
    for (int i=0;i<8;i++){ sum += c[i]; sq += c[i]*c[i]; }
    __syncthreads();             // WAR: tile buffer reused two iterations later
  }
  sum += __shfl_xor(sum, 16, 32);
  sq  += __shfl_xor(sq,  16, 32);
  if (lane < 16){
    atomicAdd(&ssum[osub*16+lane], sum);
    atomicAdd(&ssq [osub*16+lane], sq);
  }
  __syncthreads();
  if (tid < 64){
    atomicAdd(&stats[2*(ot*64+tid)],   ssum[tid]);
    atomicAdd(&stats[2*(ot*64+tid)+1], ssq [tid]);
  }
}

// ================= c3 GEMM: BN + lrelu + max over N (WMMA, async A tiles) ========
__global__ void c3_final(const _Float16* __restrict__ xp, const _Float16* __restrict__ wh,
                         const float* __restrict__ ms, const float* __restrict__ g3,
                         const float* __restrict__ b3, float* __restrict__ xg){
  __shared__ _Float16 sfA[2][32*128];
  __shared__ float smax[128];
  const int tid = threadIdx.x;
  const int b = blockIdx.y, ot = blockIdx.x;
  const int w = tid >> 5, lane = tid & 31;
  const int osub = w & 3, nh = w >> 2;
  const int r = lane & 15, h = lane >> 4;
  const int o = ot*64 + osub*16 + r;
  const float mean = ms[2*o], inv = ms[2*o+1], gg = g3[o], bb = b3[o];
  const int lrow = tid >> 3, lch = tid & 7;
  const _Float16* gbase = xp + (size_t)b*Np*128;
  h16 bf[4];
  #pragma unroll
  for (int kk=0;kk<4;kk++)
    bf[kk] = *(const h16*)(wh + (size_t)o*128 + kk*32 + h*16);
  f8 macc = {-3.0e38f,-3.0e38f,-3.0e38f,-3.0e38f,-3.0e38f,-3.0e38f,-3.0e38f,-3.0e38f};
  constexpr int NT = Np/32;
  {
    const _Float16* gp = gbase + (size_t)lrow*128 + lch*8;
    _Float16* lp = &sfA[0][lrow*128 + lch*8];
    async_b128(gp, lp); async_b128(gp + 64, lp + 64);
  }
  for (int t=0;t<NT;t++){
    if (t+1 < NT){
      const _Float16* gp = gbase + (size_t)((t+1)*32 + lrow)*128 + lch*8;
      _Float16* lp = &sfA[(t+1)&1][lrow*128 + lch*8];
      async_b128(gp, lp); async_b128(gp + 64, lp + 64);
      wait_async_le2();
    } else {
      wait_async_le0();
    }
    __syncthreads();
    const _Float16* tile = sfA[t&1];
    f8 c = {0.f,0.f,0.f,0.f,0.f,0.f,0.f,0.f};
    #pragma unroll
    for (int kk=0;kk<4;kk++){
      h16 av = load_a_frag(tile + (nh*16 + r)*128 + kk*32, h);
      c = __builtin_amdgcn_wmma_f32_16x16x32_f16(false, av, false, bf[kk],
                                                 (short)0, c, false, false);
    }
    #pragma unroll
    for (int i=0;i<8;i++){
      float y = (c[i]-mean)*inv*gg + bb;
      y = y >= 0.f ? y : 0.2f*y;
      macc[i] = fmaxf(macc[i], y);
    }
    __syncthreads();
  }
  float m = macc[0];
  #pragma unroll
  for (int i=1;i<8;i++) m = fmaxf(m, macc[i]);
  m = fmaxf(m, __shfl_xor(m, 16, 32));
  if (lane < 16) smax[w*16+lane] = m;
  __syncthreads();
  if (tid < 64)
    xg[(size_t)b*EMBD + ot*64 + tid] = fmaxf(smax[tid], smax[64+tid]);
}

// ================= Q projection (WMMA): Q[b][n][a] = (wq/8) @ x_per =================
__global__ void q_gemm(const _Float16* __restrict__ xp, const _Float16* __restrict__ wqh,
                       float* __restrict__ Qf){
  const int b = blockIdx.y;
  const int nc = blockIdx.x*32;
  const int w = threadIdx.x >> 5, lane = threadIdx.x & 31;
  const int osub = w & 3, nh = w >> 2;
  const int r = lane & 15, h = lane >> 4;
  h16 bf[4];
  #pragma unroll
  for (int kk=0;kk<4;kk++)
    bf[kk] = *(const h16*)(wqh + ((size_t)(osub*16 + r))*128 + kk*32 + h*16);
  const int n = nc + nh*16 + r;
  const _Float16* ap = xp + ((size_t)(b*Np + n))*128;
  f8 c = {0.f,0.f,0.f,0.f,0.f,0.f,0.f,0.f};
  #pragma unroll
  for (int kk=0;kk<4;kk++){
    h16 av = load_a_frag(ap + kk*32, h);
    c = __builtin_amdgcn_wmma_f32_16x16x32_f16(false, av, false, bf[kk],
                                               (short)0, c, false, false);
  }
  const int acol = osub*16 + r;
  #pragma unroll
  for (int i=0;i<8;i++){
    const int nrow = nc + nh*16 + i + h*8;
    Qf[((size_t)(b*Np + nrow))*ATTD + acol] = c[i];
  }
}

// ================= mu / log_var / z =================
__global__ void muvar_z(const float* __restrict__ xg, const float* __restrict__ wmu,
                        const float* __restrict__ bmu, const float* __restrict__ wvar,
                        const float* __restrict__ bvar, const float* __restrict__ eps,
                        float* __restrict__ out, float* __restrict__ z){
  __shared__ float sx[EMBD];
  const int b = blockIdx.x;
  for (int i=threadIdx.x;i<EMBD;i+=blockDim.x) sx[i] = xg[(size_t)b*EMBD + i];
  __syncthreads();
  const int o = threadIdx.x;   // blockDim == 128
  float am = bmu[o], av = bvar[o];
  for (int i=0;i<EMBD;i++){
    am += wmu[(size_t)o*EMBD+i]*sx[i];
    av += wvar[(size_t)o*EMBD+i]*sx[i];
  }
  out[OUT_MU + (size_t)b*ZDIM + o] = am;
  out[OUT_LV + (size_t)b*ZDIM + o] = av;
  z[(size_t)b*ZDIM + o] = eps[(size_t)b*ZDIM + o]*expf(0.5f*av) + am;
}

// ================= cuboid decoder heads =================
__global__ void decoder(const float* __restrict__ z, const float* __restrict__ wenc,
                        const float* __restrict__ wcc1, const float* __restrict__ wcc2,
                        const float* __restrict__ wsc, const float* __restrict__ bsc,
                        const float* __restrict__ wro, const float* __restrict__ bro,
                        const float* __restrict__ wtr, const float* __restrict__ btr,
                        const float* __restrict__ we1, const float* __restrict__ be1,
                        const float* __restrict__ we2, const float* __restrict__ be2,
                        const float* __restrict__ wk, float* __restrict__ heads,
                        float* __restrict__ Kcf, float* __restrict__ out){
  const int p = blockIdx.x*blockDim.x + threadIdx.x;
  if (p >= Bn*Mcub) return;
  const int b = p / Mcub, m = p % Mcub;
  float xc[ZDIM+64];
  for (int c=0;c<ZDIM;c++) xc[c] = z[(size_t)b*ZDIM + c];
  for (int c=0;c<64;c++)   xc[ZDIM+c] = lrelu(wenc[c*Mcub + m]);
  float h1[256];
  for (int o=0;o<256;o++){
    float a = 0.f;
    for (int c=0;c<ZDIM+64;c++) a += wcc1[o*(ZDIM+64)+c]*xc[c];
    h1[o] = lrelu(a);
  }
  float xq[128];
  for (int o=0;o<128;o++){
    float a = 0.f;
    for (int c=0;c<256;c++) a += wcc2[o*256+c]*h1[c];
    xq[o] = lrelu(a);
  }
  float sc[3];
  for (int d=0;d<3;d++){
    float a = bsc[d];
    for (int c=0;c<128;c++) a += wsc[d*128+c]*xq[c];
    sc[d] = 1.f/(1.f+expf(-a));
  }
  float q[4];
  for (int j=0;j<4;j++){
    float a = bro[j];
    for (int c=0;c<128;c++) a += wro[j*128+c]*xq[c];
    q[j] = a;
  }
  float nq = sqrtf(q[0]*q[0]+q[1]*q[1]+q[2]*q[2]+q[3]*q[3]);
  nq = fmaxf(nq, 1e-12f);
  for (int j=0;j<4;j++) q[j] /= nq;
  float R[9];
  R[0]=1.f-2.f*(q[2]*q[2]+q[3]*q[3]); R[1]=2.f*(q[1]*q[2]-q[0]*q[3]); R[2]=2.f*(q[1]*q[3]+q[0]*q[2]);
  R[3]=2.f*(q[1]*q[2]+q[0]*q[3]); R[4]=1.f-2.f*(q[1]*q[1]+q[3]*q[3]); R[5]=2.f*(q[2]*q[3]-q[0]*q[1]);
  R[6]=2.f*(q[1]*q[3]-q[0]*q[2]); R[7]=2.f*(q[2]*q[3]+q[0]*q[1]); R[8]=1.f-2.f*(q[1]*q[1]+q[2]*q[2]);
  float tr[3];
  for (int d=0;d<3;d++){
    float a = btr[d];
    for (int c=0;c<128;c++) a += wtr[d*128+c]*xq[c];
    tr[d] = tanhf(a);
  }
  float ex = be2[0];
  for (int i=0;i<32;i++){
    float a = be1[i];
    for (int c=0;c<128;c++) a += we1[i*128+c]*xq[c];
    ex += we2[i]*lrelu(a);
  }
  out[OUT_EXIST + (size_t)b*Mcub + m] = ex;
  for (int a=0;a<ATTD;a++){
    float v = 0.f;
    for (int c=0;c<128;c++) v += wk[a*128+c]*xq[c];
    Kcf[((size_t)(b*Mcub+m))*ATTD + a] = v;
  }
  float* hb = heads + (size_t)(b*Mcub+m)*15;
  hb[0]=sc[0]; hb[1]=sc[1]; hb[2]=sc[2];
  for (int i=0;i<9;i++) hb[3+i] = R[i];
  hb[12]=tr[0]; hb[13]=tr[1]; hb[14]=tr[2];
}

// ================= attention softmax -> assign =================
__global__ void assign_softmax(const float* __restrict__ Qf, const float* __restrict__ Kcf,
                               float* __restrict__ out){
  const int p = blockIdx.x*blockDim.x + threadIdx.x; // Bn*Np
  if (p >= Bn*Np) return;
  const int b = p / Np;
  const float* qr = Qf + (size_t)p*ATTD;
  float qv[ATTD];
  for (int a=0;a<ATTD;a++) qv[a] = qr[a];
  float lg[Mcub]; float mx = -3.0e38f;
  for (int m=0;m<Mcub;m++){
    const float* kr = Kcf + ((size_t)(b*Mcub+m))*ATTD;
    float a = 0.f;
    for (int j=0;j<ATTD;j++) a += kr[j]*qv[j];
    lg[m] = a; mx = fmaxf(mx, a);
  }
  float s = 0.f;
  for (int m=0;m<Mcub;m++){ lg[m] = expf(lg[m]-mx); s += lg[m]; }
  const float invs = 1.f/s;
  float* ao = out + OUT_ASSIGN + (size_t)p*Mcub;
  for (int m=0;m<Mcub;m++) ao[m] = lg[m]*invs;
}

// ================= pam + vertex synthesis =================
__global__ void pam_verts(const float* __restrict__ pc, const float* __restrict__ heads,
                          float* __restrict__ out){
  const int p = blockIdx.x*blockDim.x + threadIdx.x;
  if (p >= Bn*Mcub) return;
  const int b = p / Mcub, m = p % Mcub;
  const float* as = out + OUT_ASSIGN;
  float s=0.f, sx=0.f, sy=0.f, sz=0.f;
  for (int n=0;n<Np;n++){
    float a = as[((size_t)(b*Np+n))*Mcub + m];
    const float* pp = pc + ((size_t)(b*Np+n))*3;
    s += a; sx += a*pp[0]; sy += a*pp[1]; sz += a*pp[2];
  }
  const float inv = 1.f/(s + 1.f);
  float pam[3] = {sx*inv, sy*inv, sz*inv};
  const float* hb = heads + (size_t)(b*Mcub+m)*15;
  for (int v=0;v<8;v++){
    float vx = CUBEV[v][0]*hb[0];
    float vy = CUBEV[v][1]*hb[1];
    float vz = CUBEV[v][2]*hb[2];
    for (int c=0;c<3;c++){
      float val = hb[3+c*3+0]*vx + hb[3+c*3+1]*vy + hb[3+c*3+2]*vz + hb[12+c] + pam[c];
      out[OUT_VERTS + (((size_t)(b*Mcub+m))*8 + v)*3 + c] = val;
    }
  }
}

// ================= launch =================
extern "C" void kernel_launch(void* const* d_in, const int* in_sizes, int n_in,
                              void* d_out, int out_size, void* d_ws, size_t ws_size,
                              hipStream_t stream){
  (void)in_sizes; (void)n_in; (void)out_size; (void)ws_size;
  const float* pc   = (const float*)d_in[0];
  const float* eps  = (const float*)d_in[1];
  const float* c1a  = (const float*)d_in[2];
  const float* g1a  = (const float*)d_in[3];
  const float* b1a  = (const float*)d_in[4];
  const float* c1b  = (const float*)d_in[5];
  const float* g1b  = (const float*)d_in[6];
  const float* b1b  = (const float*)d_in[7];
  const float* c2a  = (const float*)d_in[8];
  const float* g2a  = (const float*)d_in[9];
  const float* b2a  = (const float*)d_in[10];
  const float* c2b  = (const float*)d_in[11];
  const float* g2b  = (const float*)d_in[12];
  const float* b2b  = (const float*)d_in[13];
  const float* c3   = (const float*)d_in[14];
  const float* g3   = (const float*)d_in[15];
  const float* b3   = (const float*)d_in[16];
  const float* wmu  = (const float*)d_in[17];
  const float* bmu  = (const float*)d_in[18];
  const float* wvar = (const float*)d_in[19];
  const float* bvar = (const float*)d_in[20];
  const float* wenc = (const float*)d_in[21];
  const float* wcc1 = (const float*)d_in[22];
  const float* wcc2 = (const float*)d_in[23];
  const float* wsc  = (const float*)d_in[24];
  const float* bsc  = (const float*)d_in[25];
  const float* wro  = (const float*)d_in[26];
  const float* bro  = (const float*)d_in[27];
  const float* wtr  = (const float*)d_in[28];
  const float* btr  = (const float*)d_in[29];
  const float* we1  = (const float*)d_in[30];
  const float* be1  = (const float*)d_in[31];
  const float* we2  = (const float*)d_in[32];
  const float* be2  = (const float*)d_in[33];
  const float* wq   = (const float*)d_in[34];
  const float* wk   = (const float*)d_in[35];

  char* wsb = (char*)d_ws;
  int*       idxb   = (int*)(wsb + OFF_IDX);
  _Float16*  xp     = (_Float16*)(wsb + OFF_XPER);
  _Float16*  c3h    = (_Float16*)(wsb + OFF_C3H);
  _Float16*  wqh    = (_Float16*)(wsb + OFF_WQH);
  _Float16*  c2ah   = (_Float16*)(wsb + OFF_C2AH);
  _Float16*  c2bh   = (_Float16*)(wsb + OFF_C2BH);
  float*     stats3 = (float*)(wsb + OFF_STATS3);
  float*     ms3    = (float*)(wsb + OFF_MS3);
  float*     statsE = (float*)(wsb + OFF_STATSE);
  float*     msE    = (float*)(wsb + OFF_MSE);
  float*     xg     = (float*)(wsb + OFF_XG);
  float*     zbuf   = (float*)(wsb + OFF_Z);
  float*     heads  = (float*)(wsb + OFF_HEADS);
  float*     Kcf    = (float*)(wsb + OFF_KC);
  float*     Qf     = (float*)(wsb + OFF_Q);
  float*     out    = (float*)d_out;

  float* st1a = statsE + 0*128; float* ms1a = msE + 0*128;
  float* st1b = statsE + 1*128; float* ms1b = msE + 1*128;
  float* st2a = statsE + 2*128; float* ms2a = msE + 2*128;
  float* st2b = statsE + 3*128; float* ms2b = msE + 3*128;

  // zero all BN stat accumulators (stats3 .. msE region)
  (void)hipMemsetAsync(wsb + OFF_STATS3, 0, OFF_XG - OFF_STATS3, stream);

  const float invPts = 1.f/(float)((long long)Bn*Np*KNN);
  const float invBN  = 1.f/(float)(Bn*Np);
  const int   ptsBlk = (Bn*Np*KNN)/256;   // 5120

  knn_kernel<<<dim3(Np/128, Bn), 128, 0, stream>>>(pc, idxb);

  cvt_half<<<(EMBD*128)/256, 256, 0, stream>>>(c3, c3h, EMBD*128, 1.0f);
  cvt_half<<<(ATTD*128)/256, 256, 0, stream>>>(wq, wqh, ATTD*128, 0.125f); // fold 1/sqrt(64)
  cvt_half<<<(64*128)/256,   256, 0, stream>>>(c2a, c2ah, 64*128, 1.0f);
  cvt_half<<<(64*64)/256,    256, 0, stream>>>(c2b, c2bh, 64*64, 1.0f);

  // EdgeConv stage 1 (CIN=6, scalar) -> x_per channels [0,64)
  ec1_stats_a<<<ptsBlk, 256, 0, stream>>>(pc, idxb, c1a, st1a);
  finalize_stats<<<1, 64, 0, stream>>>(st1a, ms1a, 64, invPts);
  ec1_stats_b<<<ptsBlk, 256, 0, stream>>>(pc, idxb, c1a, ms1a, g1a, b1a, c1b, st1b);
  finalize_stats<<<1, 64, 0, stream>>>(st1b, ms1b, 64, invPts);
  ec1_final<<<(Bn*Np)/256, 256, 0, stream>>>(pc, idxb, c1a, ms1a, g1a, b1a,
                                             c1b, ms1b, g1b, b1b, xp);

  // EdgeConv stage 2 (CIN=128, WMMA) -> x_per channels [64,128)
  ec2_wmma<0><<<Bn*Np, 256, 0, stream>>>(xp, idxb, c2ah, c2bh,
                                         ms2a, g2a, b2a, ms2b, g2b, b2b, st2a, xp);
  finalize_stats<<<1, 64, 0, stream>>>(st2a, ms2a, 64, invPts);
  ec2_wmma<1><<<Bn*Np, 256, 0, stream>>>(xp, idxb, c2ah, c2bh,
                                         ms2a, g2a, b2a, ms2b, g2b, b2b, st2b, xp);
  finalize_stats<<<1, 64, 0, stream>>>(st2b, ms2b, 64, invPts);
  ec2_wmma<2><<<Bn*Np, 256, 0, stream>>>(xp, idxb, c2ah, c2bh,
                                         ms2a, g2a, b2a, ms2b, g2b, b2b, st2b, xp);

  // c3: WMMA two-pass (stats, then BN+lrelu+max fused), async-staged A tiles
  c3_stats<<<dim3(EMBD/64, Bn), 256, 0, stream>>>(xp, c3h, stats3);
  finalize_stats<<<EMBD/64, 64, 0, stream>>>(stats3, ms3, EMBD, invBN);
  c3_final<<<dim3(EMBD/64, Bn), 256, 0, stream>>>(xp, c3h, ms3, g3, b3, xg);

  // VAE head
  muvar_z<<<Bn, 128, 0, stream>>>(xg, wmu, bmu, wvar, bvar, eps, out, zbuf);

  // cuboid decoder
  decoder<<<(Bn*Mcub+127)/128, 128, 0, stream>>>(zbuf, wenc, wcc1, wcc2, wsc, bsc,
                                                 wro, bro, wtr, btr, we1, be1,
                                                 we2, be2, wk, heads, Kcf, out);

  // attention
  q_gemm<<<dim3(Np/32, Bn), 256, 0, stream>>>(xp, wqh, Qf);
  assign_softmax<<<(Bn*Np)/256, 256, 0, stream>>>(Qf, Kcf, out);

  // pam + vertices
  pam_verts<<<(Bn*Mcub+127)/128, 128, 0, stream>>>(pc, heads, out);
}
